// AudioGuidedVisualAttn_56118042689758
// MI455X (gfx1250) — compile-verified, matrix operand order checked by reading, not run on hardware
//
#include <hip/hip_runtime.h>
#include <hip/hip_bf16.h>

// ---------------------------------------------------------------------------
// Problem constants (B=16, T=60 -> F=960 frames, N=196 patches, C=512, H=4, D=128)
// ---------------------------------------------------------------------------
#define FRAMES 960
#define NPATCH 196
#define CDIM   512
#define HEADS  4
#define DHEAD  128

typedef __attribute__((ext_vector_type(16))) __bf16 v16bf;
typedef __attribute__((ext_vector_type(8)))  float  v8f;

union BF16x16 { v16bf v; __bf16 h[16]; unsigned short u[16]; };

// Load 8 consecutive fp32 (32B aligned) and convert to 8 bf16 (native v_cvt).
__device__ __forceinline__ void load8_bf(const float* __restrict__ p, __bf16* dst) {
    const float4* p4 = (const float4*)p;
    float4 a = p4[0];
    float4 b = p4[1];
    dst[0] = (__bf16)a.x; dst[1] = (__bf16)a.y; dst[2] = (__bf16)a.z; dst[3] = (__bf16)a.w;
    dst[4] = (__bf16)b.x; dst[5] = (__bf16)b.y; dst[6] = (__bf16)b.z; dst[7] = (__bf16)b.w;
}

// ---------------------------------------------------------------------------
// Generic WMMA bf16 GEMM:  Out[m,j] = act( sum_k A[m,k] * W[j*wrs + k*wcs]
//                                          + bias[j] ) + residual[m,j]
//   wrs/wcs select A@W^T (wrs=ld, wcs=1) or A@W (wrs=1, wcs=ld).
// One 16x16 output tile per wave32; 8 waves per block.
// ---------------------------------------------------------------------------
__global__ __launch_bounds__(256) void wmma_gemm_bf16_kernel(
    const float* __restrict__ A, int lda,
    const float* __restrict__ W, int wrs, int wcs,
    const float* __restrict__ bias,
    const float* __restrict__ residual, int ldr,
    float* __restrict__ Out, int ldo,
    int M, int N, int K, int relu)
{
    const int tn = (N + 15) >> 4;
    const int tm = (M + 15) >> 4;
    const int tiles = tm * tn;
    const int tileId = blockIdx.x * 8 + (threadIdx.x >> 5);
    if (tileId >= tiles) return;

    const int ti = tileId / tn;
    const int tj = tileId - ti * tn;
    const int m0 = ti << 4;
    const int j0 = tj << 4;

    const int lane = threadIdx.x & 31;
    const int kb   = (lane >> 4) * 8;      // K sub-block select per ISA A/B layout

    int mr = m0 + (lane & 15);
    if (mr > M - 1) mr = M - 1;            // clamp (tail rows recomputed, not stored)
    const float* arow = A + (size_t)mr * lda;

    const int  j  = j0 + (lane & 15);
    const bool jv = (j < N);
    const float* wj = W + (size_t)(jv ? j : 0) * wrs;

    v8f acc = {};
    for (int k0 = 0; k0 < K; k0 += 32) {
        BF16x16 af, bf;
        load8_bf(arow + k0 + kb,      &af.h[0]);
        load8_bf(arow + k0 + kb + 16, &af.h[8]);
        if (wcs == 1) {
            load8_bf(wj + k0 + kb,      &bf.h[0]);
            load8_bf(wj + k0 + kb + 16, &bf.h[8]);
        } else {
            // strided W access: base + e*wcs folds into load immediate offsets
            const float* wb = wj + (size_t)(k0 + kb) * wcs;
            #pragma unroll
            for (int e = 0; e < 8; ++e) {
                bf.h[e]     = (__bf16)wb[e * wcs];
                bf.h[8 + e] = (__bf16)wb[(e + 16) * wcs];
            }
        }
        if (!jv) {
            #pragma unroll
            for (int e = 0; e < 16; ++e) bf.u[e] = 0;
        }
        acc = __builtin_amdgcn_wmma_f32_16x16x32_bf16(
                  false, af.v, false, bf.v, (short)0, acc, false, false);
    }

    const float bj = (bias && jv) ? bias[j] : 0.f;
    #pragma unroll
    for (int r = 0; r < 8; ++r) {
        const int row = m0 + ((lane < 16) ? r : (r + 8));   // C layout: VGPR r -> M=r / r+8
        if (row < M && jv) {
            float v = acc[r] + bj;
            if (relu) v = fmaxf(v, 0.f);
            if (residual) v += residual[(size_t)row * ldr + j];
            Out[(size_t)row * ldo + j] = v;
        }
    }
}

// ---------------------------------------------------------------------------
// Per-frame fused attention (algebraically collapsed):
//   phase 1: scores[n,h] = scale * (P[f,h,:] . X[n,:]) + q_h.bk_h   (WMMA)
//   phase 2: softmax over n per head                                (wave32 shfl)
//   phase 3: y[f,h,:]   = sum_n w[n,h] * X[n,:]                     (WMMA)
// X (visual_patch_feat) is streamed from HBM; this kernel is the bandwidth-
// bound hot path (2 x 385 MB fp32 ~ 33 us at 23.3 TB/s).
// ---------------------------------------------------------------------------
__global__ __launch_bounds__(256) void attn_frame_kernel(
    const float* __restrict__ visual,     // [F, N, C]
    const float* __restrict__ Pg,         // [F, H, C]
    const float* __restrict__ qg,         // [F, C]
    const float* __restrict__ in_proj_b,  // [3C]
    float* __restrict__ Yg)               // [F, H, C]
{
    const int f = blockIdx.x;
    const float* Xf = visual + (size_t)f * (NPATCH * CDIM);

    __shared__ float  s_sc[208][4];        // padded scores (13 n-tiles)
    __shared__ __bf16 s_w[224][4];         // bf16 softmax weights, zero-padded
    __shared__ float  s_cb[4];

    const int lane = threadIdx.x & 31;
    const int wave = threadIdx.x >> 5;
    const int kb   = (lane >> 4) * 8;

    // ---- phase 0: zero pads, compute score bias cb[h] = q_h . bk_h ----
    if (threadIdx.x < 4) s_cb[threadIdx.x] = 0.f;
    for (int i = threadIdx.x; i < 224 * 4; i += 256) ((unsigned short*)s_w)[i] = 0;
    __syncthreads();
    {
        const int h  = threadIdx.x >> 6;          // 0..3
        const int d0 = (threadIdx.x & 63) * 2;    // 0..126
        const float* qf = qg + (size_t)f * CDIM + h * DHEAD;
        const float* bk = in_proj_b + CDIM + h * DHEAD;
        atomicAdd(&s_cb[h], qf[d0] * bk[d0] + qf[d0 + 1] * bk[d0 + 1]);
    }
    __syncthreads();

    // ---- phase 1: scores via WMMA (A = X rows, B = P padded to 16 head cols) ----
    const int j = lane & 15;                       // head column (valid < 4)
    const float* brow = Pg + (size_t)f * (HEADS * CDIM) + j * CDIM;
    for (int t = wave; t < 13; t += 8) {
        const int n0 = t << 4;
        int n = n0 + (lane & 15);
        if (n > NPATCH - 1) n = NPATCH - 1;        // clamp (ignored rows)
        const float* arow = Xf + (size_t)n * CDIM;
        v8f acc = {};
        for (int k0 = 0; k0 < CDIM; k0 += 32) {
            if (k0 + 96 < CDIM) __builtin_prefetch(arow + k0 + 96, 0, 1);
            BF16x16 af, bf;
            load8_bf(arow + k0 + kb,      &af.h[0]);
            load8_bf(arow + k0 + kb + 16, &af.h[8]);
            if (j < HEADS) {
                load8_bf(brow + k0 + kb,      &bf.h[0]);
                load8_bf(brow + k0 + kb + 16, &bf.h[8]);
            } else {
                #pragma unroll
                for (int e = 0; e < 16; ++e) bf.u[e] = 0;
            }
            acc = __builtin_amdgcn_wmma_f32_16x16x32_bf16(
                      false, af.v, false, bf.v, (short)0, acc, false, false);
        }
        if (j < HEADS) {
            #pragma unroll
            for (int r = 0; r < 8; ++r) {
                const int row = n0 + ((lane < 16) ? r : (r + 8));
                s_sc[row][j] = acc[r] * 0.08838834764831845f + s_cb[j]; // 1/sqrt(128)
            }
        }
    }
    __syncthreads();

    // ---- phase 2: softmax over n (one head per wave, wave32 reductions) ----
    if (wave < HEADS) {
        const int h = wave;
        float mx = -1e30f;
        for (int n = lane; n < NPATCH; n += 32) mx = fmaxf(mx, s_sc[n][h]);
        #pragma unroll
        for (int off = 16; off; off >>= 1) mx = fmaxf(mx, __shfl_xor(mx, off, 32));
        float sum = 0.f;
        for (int n = lane; n < NPATCH; n += 32) {
            float e = __expf(s_sc[n][h] - mx);
            s_sc[n][h] = e;
            sum += e;
        }
        #pragma unroll
        for (int off = 16; off; off >>= 1) sum += __shfl_xor(sum, off, 32);
        const float inv = 1.f / sum;
        for (int n = lane; n < NPATCH; n += 32) s_w[n][h] = (__bf16)(s_sc[n][h] * inv);
    }
    __syncthreads();

    // ---- phase 3: y[h,:] = W @ X via WMMA (A = weights 16x224 padded) ----
    const int cl = lane & 15;
    const int h  = lane & 15;                      // A row = head (valid < 4)
    const __bf16 z = (__bf16)0.f;
    for (int ct = wave; ct < 32; ct += 8) {
        const int c0 = ct << 4;
        v8f acc = {};
        for (int nc = 0; nc < 7; ++nc) {
            const int nb = nc * 32;
            BF16x16 af, bf;
            // base pointers: e strides are compile-time constants (fold to offsets)
            const float*  xb = Xf + (size_t)(nb + kb) * CDIM + c0 + cl;
            const __bf16* wb = &s_w[nb + kb][h];
            #pragma unroll
            for (int e = 0; e < 8; ++e) {
                const int n1 = nb + kb + e;
                const int n2 = n1 + 16;
                af.h[e]     = (h < HEADS) ? wb[e * 4]        : z;
                af.h[8 + e] = (h < HEADS) ? wb[(e + 16) * 4] : z;
                bf.h[e]     = (n1 < NPATCH) ? (__bf16)xb[e * CDIM]        : z;
                bf.h[8 + e] = (n2 < NPATCH) ? (__bf16)xb[(e + 16) * CDIM] : z;
            }
            acc = __builtin_amdgcn_wmma_f32_16x16x32_bf16(
                      false, af.v, false, bf.v, (short)0, acc, false, false);
        }
        if (lane < 16) {
            #pragma unroll
            for (int r = 0; r < HEADS; ++r)        // rows 0..3 = heads
                Yg[(size_t)f * (HEADS * CDIM) + r * CDIM + c0 + cl] = acc[r];
        }
    }
}

// ---------------------------------------------------------------------------
// Row LayerNorm (population variance, eps=1e-5): out = (x-mu)*rsqrt(var)*g + b
// ---------------------------------------------------------------------------
__global__ __launch_bounds__(256) void layernorm_kernel(
    const float* __restrict__ X, const float* __restrict__ g,
    const float* __restrict__ b, float* __restrict__ out)
{
    const int row = blockIdx.x;
    const float* x = X + (size_t)row * CDIM;
    float s = 0.f, ss = 0.f;
    for (int c = threadIdx.x; c < CDIM; c += 256) { float v = x[c]; s += v; ss += v * v; }
    #pragma unroll
    for (int off = 16; off; off >>= 1) { s += __shfl_xor(s, off, 32); ss += __shfl_xor(ss, off, 32); }
    __shared__ float rs[8], rss[8];
    const int wid = threadIdx.x >> 5;
    if ((threadIdx.x & 31) == 0) { rs[wid] = s; rss[wid] = ss; }
    __syncthreads();
    if (threadIdx.x == 0) {
        float a = 0.f, bb = 0.f;
        for (int i = 0; i < 8; ++i) { a += rs[i]; bb += rss[i]; }
        rs[0] = a; rss[0] = bb;
    }
    __syncthreads();
    const float mean = rs[0] * (1.f / CDIM);
    const float var  = rss[0] * (1.f / CDIM) - mean * mean;
    const float inv  = rsqrtf(var + 1e-5f);
    for (int c = threadIdx.x; c < CDIM; c += 256)
        out[(size_t)row * CDIM + c] = (x[c] - mean) * inv * g[c] + b[c];
}

// ---------------------------------------------------------------------------
static void gemm(const float* A, int lda, const float* W, int wrs, int wcs,
                 const float* bias, const float* res, int ldr,
                 float* Out, int ldo, int M, int N, int K, int relu,
                 hipStream_t stream)
{
    const int tiles  = ((M + 15) / 16) * ((N + 15) / 16);
    const int blocks = (tiles + 7) / 8;
    wmma_gemm_bf16_kernel<<<blocks, 256, 0, stream>>>(
        A, lda, W, wrs, wcs, bias, res, ldr, Out, ldo, M, N, K, relu);
}

extern "C" void kernel_launch(void* const* d_in, const int* in_sizes, int n_in,
                              void* d_out, int out_size, void* d_ws, size_t ws_size,
                              hipStream_t stream)
{
    const float* audio  = (const float*)d_in[0];   // [960, 512]
    const float* visual = (const float*)d_in[1];   // [960, 196, 512]
    const float* in_w   = (const float*)d_in[2];   // [1536, 512]
    const float* in_b   = (const float*)d_in[3];   // [1536]
    const float* op_w   = (const float*)d_in[4];   // [512, 512]
    const float* op_b   = (const float*)d_in[5];
    const float* l1_w   = (const float*)d_in[6];
    const float* l1_b   = (const float*)d_in[7];
    const float* l2_w   = (const float*)d_in[8];
    const float* l2_b   = (const float*)d_in[9];
    const float* lng    = (const float*)d_in[10];
    const float* lnb    = (const float*)d_in[11];
    float* out = (float*)d_out;

    float* ws   = (float*)d_ws;
    float* qf   = ws;                          // [960, 512]
    float* Pg   = qf   + FRAMES * CDIM;        // [960, 4, 512]
    float* Yg   = Pg   + FRAMES * HEADS * CDIM;// [960, 4, 512]
    float* apre = Yg   + FRAMES * HEADS * CDIM;// [960, 512]
    float* aout = apre + FRAMES * CDIM;        // [960, 512]
    float* hid  = aout + FRAMES * CDIM;        // [960, 512]
    float* xres = hid  + FRAMES * CDIM;        // [960, 512]

    // 1) q = audio @ Wq^T + bq
    gemm(audio, CDIM, in_w, CDIM, 1, in_b, nullptr, 0,
         qf, CDIM, FRAMES, CDIM, CDIM, 0, stream);

    // 2) P_h = q_h @ Wk_h   (collapses the entire K projection)
    for (int h = 0; h < HEADS; ++h)
        gemm(qf + h * DHEAD, CDIM,
             in_w + (size_t)(CDIM + h * DHEAD) * CDIM, 1, CDIM,
             nullptr, nullptr, 0,
             Pg + h * CDIM, HEADS * CDIM,
             FRAMES, CDIM, DHEAD, 0, stream);

    // 3) fused per-frame attention: scores -> softmax -> weighted visual mean y
    attn_frame_kernel<<<FRAMES, 256, 0, stream>>>(visual, Pg, qf, in_b, Yg);

    // 4) attn_pre_h = y_h @ Wv_h^T + bv_h  (collapses the V projection)
    for (int h = 0; h < HEADS; ++h)
        gemm(Yg + h * CDIM, HEADS * CDIM,
             in_w + (size_t)(2 * CDIM + h * DHEAD) * CDIM, CDIM, 1,
             in_b + 2 * CDIM + h * DHEAD, nullptr, 0,
             apre + h * DHEAD, CDIM,
             FRAMES, DHEAD, CDIM, 0, stream);

    // 5) out_proj, 6) FFN lin1+ReLU, 7) lin2 + residual
    gemm(apre, CDIM, op_w, CDIM, 1, op_b, nullptr, 0,
         aout, CDIM, FRAMES, CDIM, CDIM, 0, stream);
    gemm(aout, CDIM, l1_w, CDIM, 1, l1_b, nullptr, 0,
         hid, CDIM, FRAMES, CDIM, CDIM, 1, stream);
    gemm(hid, CDIM, l2_w, CDIM, 1, l2_b, aout, CDIM,
         xres, CDIM, FRAMES, CDIM, CDIM, 0, stream);

    // 8) LayerNorm -> d_out
    layernorm_kernel<<<FRAMES, 256, 0, stream>>>(xres, lng, lnb, out);
}
